// VectorQuantizer_12128987644473
// MI455X (gfx1250) — matrix operand order repaired and checked by reading, not compiled, not run on hardware
//
#include <hip/hip_runtime.h>

typedef __attribute__((ext_vector_type(16))) _Float16 v16h;
typedef __attribute__((ext_vector_type(8)))  _Float16 v8h;
typedef __attribute__((ext_vector_type(8)))  float    v8f;

#define EPSF 1e-12f

constexpr int N_ = 32, L_ = 2048, D_ = 256, K_ = 1024;
constexpr int ROWS = N_ * L_;        // 65536
constexpr int NLD  = ROWS * D_;      // output elements before loss scalar
constexpr int CHUNK  = 32;           // codes staged in LDS per chunk
constexpr int NCHUNK = K_ / CHUNK;   // 32
constexpr int PITCH  = 264;          // halves per staged row (4-bank lane stride -> conflict-free b128)

// ---- LDS byte-offset of a __shared__ object (generic -> AS(3) -> u32) ----
template <typename T>
__device__ __forceinline__ uint32_t lds_addr(T* p) {
  return (uint32_t)(uintptr_t)(__attribute__((address_space(3))) T*)p;
}
__device__ __forceinline__ void wait_async0() {
  asm volatile("s_wait_asynccnt 0" ::: "memory");
}

// ---------------- Kernel 1: codebook prep + accumulator zeroing ----------------
__global__ void __launch_bounds__(256) vq_prep(const float* __restrict__ cb,
                                               _Float16* __restrict__ wh,
                                               float* __restrict__ wq,
                                               float* __restrict__ ck,
                                               float* __restrict__ acc) {
  const int k = blockIdx.x, t = threadIdx.x;
  const float v = cb[(size_t)k * D_ + t];
  float s = v * v;
  #pragma unroll
  for (int off = 16; off; off >>= 1) s += __shfl_xor(s, off, 32);
  __shared__ float ls[8];
  __shared__ float tot;
  const int lane = t & 31, w = t >> 5;
  if (lane == 0) ls[w] = s;
  __syncthreads();
  if (t == 0) { float a = 0.f; for (int i = 0; i < 8; i++) a += ls[i]; tot = a; }
  __syncthreads();
  const float ssq  = tot;
  const float sc   = 1.0f / fmaxf(sqrtf(ssq), EPSF);
  const float wn   = v * sc;                 // normalized codebook element
  const float wnsq = ssq * sc * sc;          // ||wn||^2  (~1)
  const float qsc  = 1.0f / fmaxf(sqrtf(wnsq), EPSF);
  wh[(size_t)k * D_ + t] = (_Float16)wn;     // f16 copy for WMMA B operand
  wq[(size_t)k * D_ + t] = wn * qsc;         // pre-renormalized gather row
  if (t == 0) ck[k] = wnsq;
  if (k == 0 && t == 0) { acc[0] = 0.f; acc[1] = 0.f; }
}

// ---------------- Kernel 2: fused normalize + WMMA argmin + gather + loss ----------------
__global__ void __launch_bounds__(256) vq_main(const float* __restrict__ x,
                                               const int* __restrict__ mask,
                                               const _Float16* __restrict__ wh,
                                               const float* __restrict__ wq,
                                               const float* __restrict__ ck,
                                               float* __restrict__ out,
                                               float* __restrict__ acc) {
  __shared__ _Float16 lb[2][CHUNK * PITCH];  // double-buffered staged codebook (33 KB)
  __shared__ float    sck[2][CHUNK];         // staged ||w||^2
  __shared__ int      sIdx[8][32];

  const int t    = threadIdx.x;
  const int lane = t & 31;
  const int wv   = t >> 5;                   // 8 waves per block
  const int tile = blockIdx.x * 8 + wv;      // 0..2047; 32 x-rows per wave
  const int row0 = tile * 32;
  const int r    = lane & 15;                // local row / local code slot
  const int hi   = lane >> 4;                // half-wave select

  // ---- async DMA stage of one codebook chunk into LDS buffer `sel` ----
  auto stage = [&](int sel, int cc) {
    const int srow = t >> 3, p = t & 7;      // 32 rows x 8 x 64B segments
    const uint32_t ldst = lds_addr(&lb[sel][srow * PITCH + p * 32]);
    const uint64_t ga   = (uint64_t)(uintptr_t)(wh + (size_t)(cc + srow) * D_ + p * 32);
    asm volatile("global_load_async_to_lds_b128 %0, %1, off"           :: "v"(ldst), "v"(ga) : "memory");
    asm volatile("global_load_async_to_lds_b128 %0, %1, off offset:16" :: "v"(ldst), "v"(ga) : "memory");
    asm volatile("global_load_async_to_lds_b128 %0, %1, off offset:32" :: "v"(ldst), "v"(ga) : "memory");
    asm volatile("global_load_async_to_lds_b128 %0, %1, off offset:48" :: "v"(ldst), "v"(ga) : "memory");
    if (t < CHUNK) {
      const uint32_t ld2 = lds_addr(&sck[sel][t]);
      const uint64_t ga2 = (uint64_t)(uintptr_t)(ck + cc + t);
      asm volatile("global_load_async_to_lds_b32 %0, %1, off" :: "v"(ld2), "v"(ga2) : "memory");
    }
  };

  // ---- pass 1: squared norms of this lane-pair's two rows ----
  const float4* xr0 = (const float4*)(x + (size_t)(row0 + r) * D_ + hi * 128);
  const float4* xr1 = (const float4*)(x + (size_t)(row0 + 16 + r) * D_ + hi * 128);
  float ssq0 = 0.f, ssq1 = 0.f;
  #pragma unroll
  for (int i = 0; i < 32; i++) {
    float4 q0 = xr0[i], q1 = xr1[i];
    ssq0 += q0.x * q0.x + q0.y * q0.y + q0.z * q0.z + q0.w * q0.w;
    ssq1 += q1.x * q1.x + q1.y * q1.y + q1.z * q1.z + q1.w * q1.w;
  }
  ssq0 += __shfl_xor(ssq0, 16, 32);
  ssq1 += __shfl_xor(ssq1, 16, 32);
  const float sc0 = 1.0f / fmaxf(sqrtf(ssq0), EPSF);
  const float sc1 = 1.0f / fmaxf(sqrtf(ssq1), EPSF);

  stage(0, 0);                                // kick off first DMA early

  // ---- pass 2: build the two A operands in the ISA 16-bit A layout ----
  v16h a0[8], a1[8];
  const float* xw0 = x + (size_t)(row0 + r) * D_;
  const float* xw1 = x + (size_t)(row0 + 16 + r) * D_;
  #pragma unroll
  for (int j = 0; j < 8; j++) {
    const int kb = j * 32 + hi * 8;
    #pragma unroll
    for (int e = 0; e < 8; e++) {
      a0[j][e]     = (_Float16)(xw0[kb + e] * sc0);
      a0[j][8 + e] = (_Float16)(xw0[kb + 16 + e] * sc0);
      a1[j][e]     = (_Float16)(xw1[kb + e] * sc1);
      a1[j][8 + e] = (_Float16)(xw1[kb + 16 + e] * sc1);
    }
  }

  // ---- argmax of (2*dot - ||w||^2) over K codes, double-buffered chunks ----
  float best0[8], best1[8];
  int   bidx0[8], bidx1[8];
  #pragma unroll
  for (int j = 0; j < 8; j++) {
    best0[j] = -3.4e38f; bidx0[j] = 0;
    best1[j] = -3.4e38f; bidx1[j] = 0;
  }

  for (int ci = 0; ci < NCHUNK; ci++) {
    const int sel = ci & 1;
    const int cc  = ci * CHUNK;
    wait_async0();                            // DMA for this chunk done (this wave)
    __syncthreads();                          // publish LDS to all waves
    if (ci + 1 < NCHUNK) stage(sel ^ 1, cc + CHUNK);

    const float wsq0 = sck[sel][r];           // hoisted ||w||^2 for both code tiles
    const float wsq1 = sck[sel][16 + r];
    const _Float16* wbase = &lb[sel][hi * 8];
    auto loadB = [&](int cl, int j) -> v16h {
      union { v16h v; v8h h[2]; } b;          // 16B-aligned LDS reads
      const _Float16* p = wbase + cl * PITCH + j * 32;
      b.h[0] = *(const v8h*)(p);
      b.h[1] = *(const v8h*)(p + 16);
      return b.v;
    };

    // flattened 16-step (code-tile, j) WMMA pipeline with 1-step B lookahead
    v8f c00 = {}, c01 = {}, c10 = {}, c11 = {};
    v16h bc = loadB(r, 0);
    #pragma unroll
    for (int j = 0; j < 8; j++) {
      v16h bn = (j < 7) ? loadB(r, j + 1) : loadB(16 + r, 0);
      c00 = __builtin_amdgcn_wmma_f32_16x16x32_f16(false, a0[j], false, bc,
                                                   (short)0, c00, false, false);
      c01 = __builtin_amdgcn_wmma_f32_16x16x32_f16(false, a1[j], false, bc,
                                                   (short)0, c01, false, false);
      bc = bn;
    }
    #pragma unroll
    for (int j = 0; j < 8; j++) {
      v16h bn = bc;
      if (j < 7) bn = loadB(16 + r, j + 1);
      c10 = __builtin_amdgcn_wmma_f32_16x16x32_f16(false, a0[j], false, bc,
                                                   (short)0, c10, false, false);
      c11 = __builtin_amdgcn_wmma_f32_16x16x32_f16(false, a1[j], false, bc,
                                                   (short)0, c11, false, false);
      bc = bn;
    }

    // best updates: code cc+r folded before cc+16+r (preserves first-min ties)
    #pragma unroll
    for (int j = 0; j < 8; j++) {
      float s = 2.0f * c00[j] - wsq0;
      if (s > best0[j]) { best0[j] = s; bidx0[j] = cc + r; }
      s = 2.0f * c01[j] - wsq0;
      if (s > best1[j]) { best1[j] = s; bidx1[j] = cc + r; }
    }
    #pragma unroll
    for (int j = 0; j < 8; j++) {
      float s = 2.0f * c10[j] - wsq1;
      if (s > best0[j]) { best0[j] = s; bidx0[j] = cc + 16 + r; }
      s = 2.0f * c11[j] - wsq1;
      if (s > best1[j]) { best1[j] = s; bidx1[j] = cc + 16 + r; }
    }
    __syncthreads();                          // all waves done reading lb[sel]
  }

  // ---- argmax reduction across the 16 lanes of each half-wave ----
  #pragma unroll
  for (int j = 0; j < 8; j++) {
    #pragma unroll
    for (int off = 1; off < 16; off <<= 1) {
      float os = __shfl_xor(best0[j], off, 32);
      int   oi = __shfl_xor(bidx0[j], off, 32);
      if (os > best0[j] || (os == best0[j] && oi < bidx0[j])) { best0[j] = os; bidx0[j] = oi; }
      os = __shfl_xor(best1[j], off, 32);
      oi = __shfl_xor(bidx1[j], off, 32);
      if (os > best1[j] || (os == best1[j] && oi < bidx1[j])) { best1[j] = os; bidx1[j] = oi; }
    }
  }
  if (lane == 0 || lane == 16) {
    #pragma unroll
    for (int j = 0; j < 8; j++) {
      sIdx[wv][hi * 8 + j]      = bidx0[j];   // rows row0 + 8*hi + j
      sIdx[wv][16 + hi * 8 + j] = bidx1[j];   // rows row0 + 16 + 8*hi + j
    }
  }
  __syncthreads();

  // ---- output gather + fused masked SSE (two rows per lane) ----
  const int code0 = sIdx[wv][r];
  const int code1 = sIdx[wv][16 + r];
  const int m0 = mask[row0 + r] != 0;
  const int m1 = mask[row0 + 16 + r] != 0;
  const float4* q0v = (const float4*)(wq + (size_t)code0 * D_ + hi * 128);
  const float4* q1v = (const float4*)(wq + (size_t)code1 * D_ + hi * 128);
  float4* o0 = (float4*)(out + (size_t)(row0 + r) * D_ + hi * 128);
  float4* o1 = (float4*)(out + (size_t)(row0 + 16 + r) * D_ + hi * 128);
  float sse0 = 0.f, sse1 = 0.f;
  #pragma unroll
  for (int i = 0; i < 32; i++) {
    const float4 q0 = q0v[i], q1 = q1v[i];
    o0[i] = q0; o1[i] = q1;
    const float4 x0 = xr0[i], x1 = xr1[i];
    float dx = q0.x - x0.x * sc0, dy = q0.y - x0.y * sc0;
    float dz = q0.z - x0.z * sc0, dw = q0.w - x0.w * sc0;
    sse0 += dx * dx + dy * dy + dz * dz + dw * dw;
    dx = q1.x - x1.x * sc1; dy = q1.y - x1.y * sc1;
    dz = q1.z - x1.z * sc1; dw = q1.w - x1.w * sc1;
    sse1 += dx * dx + dy * dy + dz * dz + dw * dw;
  }
  float sse = (m0 ? sse0 : 0.f) + (m1 ? sse1 : 0.f);
  float cnt = (hi == 0) ? (float)(m0 + m1) : 0.0f;
  #pragma unroll
  for (int off = 16; off; off >>= 1) {
    sse += __shfl_xor(sse, off, 32);
    cnt += __shfl_xor(cnt, off, 32);
  }
  if (lane == 0) { atomicAdd(&acc[0], sse); atomicAdd(&acc[1], cnt); }
}

// ---------------- Kernel 3: loss epilogue ----------------
__global__ void vq_loss(const float* __restrict__ acc, float* __restrict__ out) {
  const float denom = fmaxf(acc[1] * (float)D_, 1.0f);
  out[NLD] = 1.25f * acc[0] / denom;   // q_latent + 0.25*e_latent (equal in forward)
}

extern "C" void kernel_launch(void* const* d_in, const int* in_sizes, int n_in,
                              void* d_out, int out_size, void* d_ws, size_t ws_size,
                              hipStream_t stream) {
  (void)in_sizes; (void)n_in; (void)out_size; (void)ws_size;
  const float* x    = (const float*)d_in[0];
  const int*   mask = (const int*)d_in[1];
  const float* cb   = (const float*)d_in[2];
  float* out = (float*)d_out;

  char* ws = (char*)d_ws;                       // ~1.6 MB used
  _Float16* wh = (_Float16*)(ws);               // f16 normalized codebook   512 KB
  float*    wq = (float*)(ws + 524288);         // renormalized f32 rows     1 MB
  float*    ck = (float*)(ws + 1572864);        // ||wn||^2 per code         4 KB
  float*    acc = (float*)(ws + 1576960);       // [sse, count]

  vq_prep<<<K_, 256, 0, stream>>>(cb, wh, wq, ck, acc);
  vq_main<<<ROWS / 32 / 8, 256, 0, stream>>>(x, mask, wh, wq, ck, out, acc);
  vq_loss<<<1, 1, 0, stream>>>(acc, out);
}
